// GroupedQueryAttention_27711128994554
// MI455X (gfx1250) — matrix operand order repaired and checked by reading, not compiled, not run on hardware
//
#include <hip/hip_runtime.h>
#include <cstdint>

// ---------------------------------------------------------------------------
// GQA attention for MI455X (gfx1250): f16 WMMA (v_wmma_f32_16x16x32_f16)
// cast(+transpose W) -> QKV GEMMs -> RoPE -> V transpose -> flash attn -> Wo
// All LDS fragment traffic is b128-contiguous (B operands pre-transposed).
// Staging uses gfx1250 async global->LDS (ASYNCcnt) when available.
// ---------------------------------------------------------------------------

typedef __attribute__((ext_vector_type(16))) _Float16 v16h;
typedef __attribute__((ext_vector_type(8)))  float    v8f;
typedef __attribute__((ext_vector_type(4)))  _Float16 h4;
typedef __attribute__((ext_vector_type(4)))  int      v4i;

#define B_SZ    2
#define S_LEN   2048
#define D_MODEL 2048
#define N_Q     32
#define N_KV    8
#define D_HEAD  64
#define GQA_G   (N_Q / N_KV)
#define SCALE   0.125f   // 64^-0.5

// ---------------- async global->LDS (gfx1250) with fallback ----------------
#if defined(__AMDGCN__) &&                                                   \
    __has_builtin(__builtin_amdgcn_global_load_async_to_lds_b128) &&         \
    __has_builtin(__builtin_amdgcn_s_wait_asynccnt)
#define ASYNC_LDS 1
#endif

__device__ __forceinline__ void cp16(void* lds, const void* g) {
#ifdef ASYNC_LDS
  __builtin_amdgcn_global_load_async_to_lds_b128(
      (__attribute__((address_space(1))) v4i*)g,
      (__attribute__((address_space(3))) v4i*)lds, 0, 0);
#else
  *(uint4*)lds = *(const uint4*)g;
#endif
}
__device__ __forceinline__ void cp_wait() {
#ifdef ASYNC_LDS
  __builtin_amdgcn_s_wait_asynccnt(0);
#endif
}

// -------------------------------- WMMA helpers ------------------------------

__device__ __forceinline__ v8f wmma_f16(v16h a, v16h b, v8f c) {
  return __builtin_amdgcn_wmma_f32_16x16x32_f16(false, a, false, b,
                                                (short)0, c, false, false);
}

// 16x32 f16 fragment from a row-major tile (A fragment; also the B fragment
// when B is stored transposed [N x K], since B(k,n) = BT[n][k]).
// ISA layout: lanes 0-15: row=lane,  K={0..7,16..23};
//             lanes 16-31: row=lane-16, K={8..15,24..31}; pairs per VGPR.
// Contiguous 16B runs -> lowers to 2x ds_load_b128 / global_load_b128.
__device__ __forceinline__ v16h load_frag_rm(const _Float16* base, int ld) {
  const int lane = threadIdx.x & 31;
  const int row  = lane & 15;
  const int khi  = (lane >> 4) << 3;
  v16h out;
#pragma unroll
  for (int i = 0; i < 8; ++i) {
    const int k = ((i & 4) << 2) + ((i & 3) << 1) + khi;
    out[2 * i]     = base[row * ld + k];
    out[2 * i + 1] = base[row * ld + k + 1];
  }
  return out;
}

// ------------------------------ cast fp32->f16 ------------------------------

__global__ __launch_bounds__(256) void cvt_f32_f16(const float* __restrict__ s,
                                                   _Float16* __restrict__ d,
                                                   int n4) {
  int i = blockIdx.x * blockDim.x + threadIdx.x;
  if (i < n4) {
    float4 f = ((const float4*)s)[i];
    h4 r = {(_Float16)f.x, (_Float16)f.y, (_Float16)f.z, (_Float16)f.w};
    ((h4*)d)[i] = r;
  }
}

// cast + transpose: W [K,N] fp32 -> WT [N,K] f16
__global__ __launch_bounds__(256) void cvt_transpose(const float* __restrict__ W,
                                                     _Float16* __restrict__ WT,
                                                     int K, int N) {
  int i = blockIdx.x * blockDim.x + threadIdx.x;
  if (i >= K * N) return;
  const int k = i / N, n = i % N;
  WT[(size_t)n * K + k] = (_Float16)W[i];
}

// V [B*S, N_KV*64] f16 -> VT [B, N_KV, 64, S] f16
__global__ __launch_bounds__(256) void transpose_v(const _Float16* __restrict__ v,
                                                   _Float16* __restrict__ vT,
                                                   int total) {
  int i = blockIdx.x * blockDim.x + threadIdx.x;
  if (i >= total) return;
  const int s  = i % S_LEN;
  const int d  = (i / S_LEN) & 63;
  const int kv = (i / (S_LEN * 64)) % N_KV;
  const int b  = i / (S_LEN * 64 * N_KV);
  vT[i] = v[(((size_t)(b * S_LEN + s)) * N_KV + kv) * D_HEAD + d];
}

// ----------------------------------- RoPE -----------------------------------

__global__ __launch_bounds__(256) void rope_kernel(_Float16* __restrict__ t,
                                                   const float* __restrict__ cosT,
                                                   const float* __restrict__ sinT,
                                                   int n, int H) {
  int i = blockIdx.x * blockDim.x + threadIdx.x;
  if (i >= n) return;
  const int d   = i & 31;
  const int h   = (i >> 5) % H;
  const int row = i / (32 * H);
  const int s   = row % S_LEN;
  _Float16* p = t + (size_t)row * (H * D_HEAD) + h * D_HEAD;
  const float c0 = cosT[s * 64 + d],      s0 = sinT[s * 64 + d];
  const float c1 = cosT[s * 64 + d + 32], s1 = sinT[s * 64 + d + 32];
  const float a = (float)p[d];
  const float b = (float)p[d + 32];
  p[d]      = (_Float16)(a * c0 - b * s0);
  p[d + 32] = (_Float16)(b * c1 + a * s1);
}

// ------------------------------- WMMA GEMM ----------------------------------
// C[M,N] = A[M,K] * BT[N,K]^T.  128 threads = 4 waves, block tile 128x64,
// wave tile 32x64 (2 A-frags x 4 B-frags = 8 WMMA per 32-K step).
template <bool STORE_F32>
__global__ __launch_bounds__(128) void gemm_wmma_kernel(
    const _Float16* __restrict__ A, const _Float16* __restrict__ BT,
    void* __restrict__ Cv, int M, int N, int K) {
  __shared__ _Float16 As[128][40];
  __shared__ _Float16 Bs[64][40];

  const int tid  = threadIdx.x;
  const int wave = tid >> 5;
  const int lane = tid & 31;
  const int m0 = blockIdx.y * 128;
  const int n0 = blockIdx.x * 64;

  v8f acc[2][4] = {};

  const int brow = tid >> 1, bseg = (tid & 1) * 16;

  for (int k0 = 0; k0 < K; k0 += 32) {
    const _Float16* ga = A  + (size_t)(m0 + tid) * K + k0;            // 32 halves
    const _Float16* gb = BT + (size_t)(n0 + brow) * K + k0 + bseg;    // 16 halves
    __builtin_prefetch((const void*)(ga + 32), 0, 1);
    __builtin_prefetch((const void*)(gb + 32), 0, 1);

    __syncthreads();                 // previous iteration reads done
    cp16(&As[tid][0], ga);
    cp16(&As[tid][8], ga + 8);
    cp16(&As[tid][16], ga + 16);
    cp16(&As[tid][24], ga + 24);
    cp16(&Bs[brow][bseg], gb);
    cp16(&Bs[brow][bseg + 8], gb + 8);
    cp_wait();
    __syncthreads();

    const v16h a0 = load_frag_rm(&As[wave * 32][0], 40);
    const v16h a1 = load_frag_rm(&As[wave * 32 + 16][0], 40);
#pragma unroll
    for (int t = 0; t < 4; ++t) {
      const v16h bf = load_frag_rm(&Bs[t * 16][0], 40);
      acc[0][t] = wmma_f16(a0, bf, acc[0][t]);
      acc[1][t] = wmma_f16(a1, bf, acc[1][t]);
    }
  }

  const int Mg = (lane >> 4) << 3;
  const int nn = lane & 15;
#pragma unroll
  for (int a = 0; a < 2; ++a) {
#pragma unroll
    for (int t = 0; t < 4; ++t) {
#pragma unroll
      for (int r = 0; r < 8; ++r) {
        const size_t gr = (size_t)(m0 + wave * 32 + a * 16 + Mg + r);
        const size_t gc = (size_t)(n0 + t * 16 + nn);
        if (STORE_F32) ((float*)Cv)[gr * N + gc] = acc[a][t][r];
        else           ((_Float16*)Cv)[gr * N + gc] = (_Float16)acc[a][t][r];
      }
    }
  }
}

// ------------------------------ flash attention -----------------------------
// One wave per (b, qhead, 32-query tile): two 16-row Q subtiles share the
// K/V staging. 16 WMMAs per 32-key chunk; online softmax per subtile.
__global__ __launch_bounds__(32) void attn_kernel(
    const _Float16* __restrict__ q, const _Float16* __restrict__ k,
    const _Float16* __restrict__ vT, _Float16* __restrict__ o) {
  const int nqt = S_LEN / 32;
  const int qt  = blockIdx.x % nqt;
  const int h   = (blockIdx.x / nqt) % N_Q;
  const int b   = blockIdx.x / (nqt * N_Q);
  const int kvh = h / GQA_G;
  const int qbase = qt * 32;
  const int lane  = threadIdx.x;

  __shared__ _Float16 Kt[32][72];   // [key][d]     -> QK^T B-frags contiguous
  __shared__ _Float16 Vt[64][40];   // [d][key]     -> PV  B-frags contiguous
  __shared__ _Float16 Pt[32][40];   // [qrow][key]  -> PV  A-frags contiguous

  v16h qa[2][2];
#pragma unroll
  for (int u = 0; u < 2; ++u) {
    const _Float16* qp =
        q + ((size_t)(b * S_LEN + qbase + u * 16)) * (N_Q * D_HEAD) + h * D_HEAD;
    qa[u][0] = load_frag_rm(qp, N_Q * D_HEAD);
    qa[u][1] = load_frag_rm(qp + 32, N_Q * D_HEAD);
  }

  v8f acc[2][4] = {};
  float mrow[2][8], lrow[2][8];
#pragma unroll
  for (int u = 0; u < 2; ++u)
#pragma unroll
    for (int r = 0; r < 8; ++r) { mrow[u][r] = -__builtin_inff(); lrow[u][r] = 0.0f; }

  const int Mg = (lane >> 4) << 3;
  const int j  = lane & 15;

  for (int kb = 0; kb < qbase + 32; kb += 32) {   // causal bound
    __syncthreads();
    // K rows: lane stages key `lane` (64 halves = 8x16B)
    const _Float16* kp =
        k + ((size_t)(b * S_LEN + kb + lane)) * (N_KV * D_HEAD) + kvh * D_HEAD;
#pragma unroll
    for (int i = 0; i < 8; ++i) cp16(&Kt[lane][i * 8], kp + i * 8);
    // V^T rows: lane stages d-rows 2*lane, 2*lane+1 (32 halves = 4x16B each)
#pragma unroll
    for (int rr = 0; rr < 2; ++rr) {
      const int d = 2 * lane + rr;
      const _Float16* vp =
          vT + ((size_t)((b * N_KV + kvh) * D_HEAD + d)) * S_LEN + kb;
      cp16(&Vt[d][0], vp);
      cp16(&Vt[d][8], vp + 8);
      cp16(&Vt[d][16], vp + 16);
      cp16(&Vt[d][24], vp + 24);
    }
    cp_wait();
    __syncthreads();

    // scores: keys kb..kb+15 (frag *0) and kb+16..kb+31 (frag *1)
    const v16h kf0a = load_frag_rm(&Kt[0][0], 72);
    const v16h kf0b = load_frag_rm(&Kt[0][32], 72);
    const v16h kf1a = load_frag_rm(&Kt[16][0], 72);
    const v16h kf1b = load_frag_rm(&Kt[16][32], 72);

#pragma unroll
    for (int u = 0; u < 2; ++u) {
      v8f s0 = {}, s1 = {};
      s0 = wmma_f16(qa[u][0], kf0a, s0);
      s0 = wmma_f16(qa[u][1], kf0b, s0);
      s1 = wmma_f16(qa[u][0], kf1a, s1);
      s1 = wmma_f16(qa[u][1], kf1b, s1);

#pragma unroll
      for (int r = 0; r < 8; ++r) {
        const int qg = qbase + u * 16 + Mg + r;
        float a0 = s0[r] * SCALE;
        float a1 = s1[r] * SCALE;
        if (kb + j > qg)      a0 = -__builtin_inff();
        if (kb + 16 + j > qg) a1 = -__builtin_inff();
        float mx = fmaxf(a0, a1);
#pragma unroll
        for (int w = 1; w < 16; w <<= 1) mx = fmaxf(mx, __shfl_xor(mx, w, 32));
        const float mnew  = fmaxf(mrow[u][r], mx);
        const float alpha = __expf(mrow[u][r] - mnew);
        const float e0 = __expf(a0 - mnew);
        const float e1 = __expf(a1 - mnew);
        float rs = e0 + e1;
#pragma unroll
        for (int w = 1; w < 16; w <<= 1) rs += __shfl_xor(rs, w, 32);
        lrow[u][r] = lrow[u][r] * alpha + rs;
        mrow[u][r] = mnew;
#pragma unroll
        for (int t = 0; t < 4; ++t) acc[u][t][r] *= alpha;
        Pt[u * 16 + Mg + r][j]      = (_Float16)e0;
        Pt[u * 16 + Mg + r][j + 16] = (_Float16)e1;
      }
    }
    __syncthreads();

    // O += P(16x32) * V(32x64), both subtiles share each V fragment
    const v16h pf0 = load_frag_rm(&Pt[0][0], 40);
    const v16h pf1 = load_frag_rm(&Pt[16][0], 40);
#pragma unroll
    for (int t = 0; t < 4; ++t) {
      const v16h vb = load_frag_rm(&Vt[t * 16][0], 40);
      acc[0][t] = wmma_f16(pf0, vb, acc[0][t]);
      acc[1][t] = wmma_f16(pf1, vb, acc[1][t]);
    }
  }

  // normalize and store f16 output [B*S, N_Q*D_HEAD]
#pragma unroll
  for (int u = 0; u < 2; ++u)
#pragma unroll
    for (int t = 0; t < 4; ++t)
#pragma unroll
      for (int r = 0; r < 8; ++r) {
        const float val = acc[u][t][r] / lrow[u][r];
        o[((size_t)(b * S_LEN + qbase + u * 16 + Mg + r)) * (N_Q * D_HEAD) +
          h * D_HEAD + t * 16 + j] = (_Float16)val;
      }
}

// --------------------------------- launcher ---------------------------------

extern "C" void kernel_launch(void* const* d_in, const int* in_sizes, int n_in,
                              void* d_out, int out_size, void* d_ws, size_t ws_size,
                              hipStream_t stream) {
  (void)in_sizes; (void)n_in; (void)out_size; (void)ws_size;

  const float* x    = (const float*)d_in[0];
  const float* ctx  = (const float*)d_in[1];
  const float* cosT = (const float*)d_in[2];
  const float* sinT = (const float*)d_in[3];
  const float* ccos = (const float*)d_in[4];
  const float* csin = (const float*)d_in[5];
  const float* Wq   = (const float*)d_in[6];
  const float* Wk   = (const float*)d_in[7];
  const float* Wv   = (const float*)d_in[8];
  const float* Wo   = (const float*)d_in[9];

  const int ROWS = B_SZ * S_LEN;                          // 4096
  const size_t nXD  = (size_t)ROWS * D_MODEL;             // 8388608
  const size_t nWq  = (size_t)D_MODEL * D_MODEL;          // 4194304
  const size_t nWkv = (size_t)D_MODEL * (N_KV * D_HEAD);  // 1048576
  const size_t nKV  = (size_t)ROWS * (N_KV * D_HEAD);     // 2097152

  _Float16* ws = (_Float16*)d_ws;
  _Float16* x16  = ws;                  // nXD
  _Float16* c16  = x16  + nXD;          // nXD
  _Float16* wqT  = c16  + nXD;          // nWq   [N,K]
  _Float16* wkT  = wqT  + nWq;          // nWkv  [N,K]
  _Float16* wvT  = wkT  + nWkv;         // nWkv  [N,K]
  _Float16* woT  = wvT  + nWkv;         // nWq   [N,K]
  _Float16* q16  = woT  + nWq;          // nXD
  _Float16* k16  = q16  + nXD;          // nKV
  _Float16* v16  = k16  + nKV;          // nKV
  _Float16* vT16 = v16  + nKV;          // nKV   [B,N_KV,64,S]
  _Float16* a16  = vT16 + nKV;          // nXD

  auto cvt = [&](const float* s, _Float16* d, size_t n) {
    int n4 = (int)(n / 4);
    cvt_f32_f16<<<(n4 + 255) / 256, 256, 0, stream>>>(s, d, n4);
  };
  cvt(x,   x16, nXD);
  cvt(ctx, c16, nXD);
  cvt_transpose<<<((int)nWq + 255) / 256, 256, 0, stream>>>(Wq, wqT, D_MODEL, D_MODEL);
  cvt_transpose<<<((int)nWkv + 255) / 256, 256, 0, stream>>>(Wk, wkT, D_MODEL, N_KV * D_HEAD);
  cvt_transpose<<<((int)nWkv + 255) / 256, 256, 0, stream>>>(Wv, wvT, D_MODEL, N_KV * D_HEAD);
  cvt_transpose<<<((int)nWq + 255) / 256, 256, 0, stream>>>(Wo, woT, D_MODEL, D_MODEL);

  // projections (B operand transposed)
  gemm_wmma_kernel<false><<<dim3(D_MODEL / 64, ROWS / 128), 128, 0, stream>>>(
      x16, wqT, q16, ROWS, D_MODEL, D_MODEL);
  gemm_wmma_kernel<false><<<dim3((N_KV * D_HEAD) / 64, ROWS / 128), 128, 0, stream>>>(
      c16, wkT, k16, ROWS, N_KV * D_HEAD, D_MODEL);
  gemm_wmma_kernel<false><<<dim3((N_KV * D_HEAD) / 64, ROWS / 128), 128, 0, stream>>>(
      c16, wvT, v16, ROWS, N_KV * D_HEAD, D_MODEL);

  // RoPE on q and k
  {
    int nq = ROWS * N_Q * 32;
    rope_kernel<<<(nq + 255) / 256, 256, 0, stream>>>(q16, cosT, sinT, nq, N_Q);
    int nk = ROWS * N_KV * 32;
    rope_kernel<<<(nk + 255) / 256, 256, 0, stream>>>(k16, ccos, csin, nk, N_KV);
  }

  // V transpose for contiguous PV fragments
  transpose_v<<<((int)nKV + 255) / 256, 256, 0, stream>>>(v16, vT16, (int)nKV);

  // attention: one wave per (b, head, 32-query tile)
  attn_kernel<<<B_SZ * N_Q * (S_LEN / 32), 32, 0, stream>>>(q16, k16, vT16, a16);

  // output projection (fp32 result)
  gemm_wmma_kernel<true><<<dim3(D_MODEL / 64, ROWS / 128), 128, 0, stream>>>(
      a16, woT, d_out, ROWS, D_MODEL, D_MODEL);
}
